// NaiveCodebook_38766374814394
// MI455X (gfx1250) — matrix operand order repaired and checked by reading, not compiled, not run on hardware
//
#include <hip/hip_runtime.h>
#include <hip/hip_bf16.h>
#include <math.h>

// ---------------------------------------------------------------------------
// Problem constants
// ---------------------------------------------------------------------------
#define BATCH   4096
#define IN_DIM  12288      // 3*64*64
#define EMBD    512
#define KBOOK   8192

typedef __attribute__((ext_vector_type(16))) __bf16 bf16x16;
typedef __attribute__((ext_vector_type(8)))  __bf16 bf16x8;
typedef __attribute__((ext_vector_type(8)))  float  f32x8;
typedef __attribute__((ext_vector_type(8)))  float  v8f;

// ---------------------------------------------------------------------------
// Workspace layout (bytes). All offsets are 32B+ aligned.
// ---------------------------------------------------------------------------
#define OFF_INPUT  ((size_t)0)                                    // f32 [4096][512]
#define OFF_WINT   (OFF_INPUT + (size_t)BATCH*EMBD*4)             // bf16 [512][12288]
#define OFF_BOOK   (OFF_WINT  + (size_t)EMBD*IN_DIM*2)            // bf16 [8192][512]
#define OFF_BNORM  (OFF_BOOK  + (size_t)KBOOK*EMBD*2)             // f32 [8192]
#define OFF_IDX    (OFF_BNORM + (size_t)KBOOK*4)                  // i32 [4096]
#define OFF_QNT    (OFF_IDX   + (size_t)BATCH*4)                  // bf16 [4096][512]
#define OFF_WOT    (OFF_QNT   + (size_t)BATCH*EMBD*2)             // bf16 [12288][512]

// ---------------------------------------------------------------------------
// Prep kernels: transpose + fp32 -> bf16 conversion of the weight matrices,
// bf16 conversion of the codebook + per-row squared norms.
// ---------------------------------------------------------------------------
__global__ __launch_bounds__(256) void prep_WinT(const float* __restrict__ Win,
                                                 __bf16* __restrict__ WT) {
    int k = blockIdx.x * 256 + threadIdx.x;   // 0..IN_DIM-1 (contiguous writes)
    int n = blockIdx.y;                       // 0..EMBD-1
    WT[(size_t)n * IN_DIM + k] = (__bf16)Win[(size_t)k * EMBD + n];
}

__global__ __launch_bounds__(256) void prep_WoutT(const float* __restrict__ Wout,
                                                  __bf16* __restrict__ WT) {
    int k = blockIdx.x * 256 + threadIdx.x;   // 0..EMBD-1
    int n = blockIdx.y;                       // 0..IN_DIM-1
    WT[(size_t)n * EMBD + k] = (__bf16)Wout[(size_t)k * IN_DIM + n];
}

__global__ __launch_bounds__(256) void prep_book(const float* __restrict__ book,
                                                 __bf16* __restrict__ bb,
                                                 float* __restrict__ bnorm) {
    int row = blockIdx.x;
    int tid = threadIdx.x;
    const float* src = book + (size_t)row * EMBD;
    float s = 0.0f;
    for (int i = tid; i < EMBD; i += 256) {
        __bf16 h = (__bf16)src[i];
        bb[(size_t)row * EMBD + i] = h;
        float f = (float)h;
        s += f * f;
    }
    #pragma unroll
    for (int m = 16; m >= 1; m >>= 1) s += __shfl_xor(s, m, 32);
    __shared__ float sm[8];
    if ((tid & 31) == 0) sm[tid >> 5] = s;
    __syncthreads();
    if (tid == 0) {
        float a = 0.0f;
        #pragma unroll
        for (int w = 0; w < 8; ++w) a += sm[w];
        bnorm[row] = a;
    }
}

// ---------------------------------------------------------------------------
// GEMM1: input[4096][512] = bf16(img1 - img2) @ bf16(W_in)   (f32 accumulate)
// One block = 16 rows x full 512 cols. 8 waves, 4 16x16 tiles per wave.
// A fragment built by fused fp32 load + subtract + cvt (images read once).
// ---------------------------------------------------------------------------
__global__ __launch_bounds__(256) void gemm1_in(const float* __restrict__ img1,
                                                const float* __restrict__ img2,
                                                const __bf16* __restrict__ WT,   // [512][12288]
                                                float* __restrict__ out) {       // [4096][512]
    const int tid  = threadIdx.x;
    const int wave = tid >> 5;
    const int lane = tid & 31;
    const int half = lane >> 4;
    const int r    = lane & 15;
    const int r0   = blockIdx.x * 16;
    const int nbase = wave * 64;

    v8f c[4] = {};
    const float* p1 = img1 + (size_t)(r0 + r) * IN_DIM + half * 8;
    const float* p2 = img2 + (size_t)(r0 + r) * IN_DIM + half * 8;

    for (int k0 = 0; k0 < IN_DIM; k0 += 32) {
        // A fragment: K = {k0+off .. +7} U {k0+off+16 .. +23}, off = half*8
        f32x8 d0 = *(const f32x8*)(p1 + k0)      - *(const f32x8*)(p2 + k0);
        f32x8 d1 = *(const f32x8*)(p1 + k0 + 16) - *(const f32x8*)(p2 + k0 + 16);
        bf16x16 a;
        #pragma unroll
        for (int i = 0; i < 8; ++i) { a[i] = (__bf16)d0[i]; a[8 + i] = (__bf16)d1[i]; }
        #pragma unroll
        for (int t = 0; t < 4; ++t) {
            // B fragment: column (nbase+t*16+r), K = {k0+half*8 .. +7} U {k0+16+half*8 .. +7}
            const __bf16* bp = WT + (size_t)(nbase + t * 16 + r) * IN_DIM + k0 + half * 8;
            bf16x8 blo = *(const bf16x8*)(bp);
            bf16x8 bhi = *(const bf16x8*)(bp + 16);
            bf16x16 b;
            #pragma unroll
            for (int i = 0; i < 8; ++i) { b[i] = blo[i]; b[8 + i] = bhi[i]; }
            c[t] = __builtin_amdgcn_wmma_f32_16x16x32_bf16(false, a, false, b,
                                                           (short)0, c[t], false, false);
        }
    }
    #pragma unroll
    for (int t = 0; t < 4; ++t) {
        int col = nbase + t * 16 + r;
        #pragma unroll
        for (int v = 0; v < 8; ++v)
            out[(size_t)(r0 + v + half * 8) * EMBD + col] = c[t][v];
    }
}

// ---------------------------------------------------------------------------
// Fused score + argmin: d2(b,k) = bnorm[k] - 2 * input[b] . book[k]  (+const)
// One block = 16 rows, scans all 8192 codebook entries with WMMA dots,
// tracks per-row running argmin, reduces via shuffles + LDS.
// ---------------------------------------------------------------------------
__global__ __launch_bounds__(256) void score_argmin(const float* __restrict__ inp,   // [4096][512]
                                                    const __bf16* __restrict__ book, // [8192][512]
                                                    const float* __restrict__ bnorm, // [8192]
                                                    int* __restrict__ indices) {     // [4096]
    const int tid  = threadIdx.x;
    const int wave = tid >> 5;
    const int lane = tid & 31;
    const int half = lane >> 4;
    const int r    = lane & 15;
    const int r0   = blockIdx.x * 16;

    float best[8];
    int   besti[8];
    #pragma unroll
    for (int v = 0; v < 8; ++v) { best[v] = 3.4e38f; besti[v] = 0; }

    const float* ap = inp + (size_t)(r0 + r) * EMBD + half * 8;

    for (int outer = 0; outer < KBOOK; outer += 512) {
        const int nbase = outer + wave * 64;
        v8f c[4] = {};
        for (int k0 = 0; k0 < EMBD; k0 += 32) {
            f32x8 d0 = *(const f32x8*)(ap + k0);
            f32x8 d1 = *(const f32x8*)(ap + k0 + 16);
            bf16x16 a;
            #pragma unroll
            for (int i = 0; i < 8; ++i) { a[i] = (__bf16)d0[i]; a[8 + i] = (__bf16)d1[i]; }
            #pragma unroll
            for (int t = 0; t < 4; ++t) {
                const __bf16* bp = book + (size_t)(nbase + t * 16 + r) * EMBD + k0 + half * 8;
                bf16x8 blo = *(const bf16x8*)(bp);
                bf16x8 bhi = *(const bf16x8*)(bp + 16);
                bf16x16 b;
                #pragma unroll
                for (int i = 0; i < 8; ++i) { b[i] = blo[i]; b[8 + i] = bhi[i]; }
                c[t] = __builtin_amdgcn_wmma_f32_16x16x32_bf16(false, a, false, b,
                                                               (short)0, c[t], false, false);
            }
        }
        #pragma unroll
        for (int t = 0; t < 4; ++t) {
            int col = nbase + t * 16 + r;
            float bn = bnorm[col];
            #pragma unroll
            for (int v = 0; v < 8; ++v) {
                float s = bn - 2.0f * c[t][v];   // row m = v + 8*half
                if (s < best[v] || (s == best[v] && col < besti[v])) {
                    best[v] = s; besti[v] = col;
                }
            }
        }
    }

    // Reduce across the 16 lanes of each half (xor bits 0..3 keeps the half).
    #pragma unroll
    for (int v = 0; v < 8; ++v) {
        #pragma unroll
        for (int m = 8; m >= 1; m >>= 1) {
            float ob = __shfl_xor(best[v], m, 32);
            int   oi = __shfl_xor(besti[v], m, 32);
            if (ob < best[v] || (ob == best[v] && oi < besti[v])) {
                best[v] = ob; besti[v] = oi;
            }
        }
    }

    __shared__ float s_best[8][2][8];
    __shared__ int   s_idx[8][2][8];
    if (r == 0) {
        #pragma unroll
        for (int v = 0; v < 8; ++v) { s_best[wave][half][v] = best[v]; s_idx[wave][half][v] = besti[v]; }
    }
    __syncthreads();
    if (tid < 16) {               // row m = tid : v = m&7, half = m>>3
        int v = tid & 7, h = tid >> 3;
        float bb = 3.4e38f; int bi = 0;
        #pragma unroll
        for (int w = 0; w < 8; ++w) {
            float x = s_best[w][h][v]; int xi = s_idx[w][h][v];
            if (x < bb || (x == bb && xi < bi)) { bb = x; bi = xi; }
        }
        indices[r0 + tid] = bi;
    }
}

// ---------------------------------------------------------------------------
// Quantize epilogue: per row, res/rand norms + scaled random vector; emit bf16.
// ---------------------------------------------------------------------------
__global__ __launch_bounds__(256) void quantize_rows(const float* __restrict__ inp,
                                                     const float* __restrict__ book_f32,
                                                     const float* __restrict__ randv,
                                                     const int* __restrict__ indices,
                                                     __bf16* __restrict__ qout) {
    const int b = blockIdx.x, tid = threadIdx.x;
    const int idx = indices[b];
    const float* x  = inp      + (size_t)b   * EMBD;
    const float* q  = book_f32 + (size_t)idx * EMBD;
    const float* rv = randv    + (size_t)b   * EMBD;

    float sres = 0.0f, srnd = 0.0f;
    for (int i = tid; i < EMBD; i += 256) {
        float d = x[i] - q[i]; sres += d * d;
        float rr = rv[i];      srnd += rr * rr;
    }
    #pragma unroll
    for (int m = 16; m >= 1; m >>= 1) {
        sres += __shfl_xor(sres, m, 32);
        srnd += __shfl_xor(srnd, m, 32);
    }
    __shared__ float sm[2][8];
    if ((tid & 31) == 0) { sm[0][tid >> 5] = sres; sm[1][tid >> 5] = srnd; }
    __syncthreads();
    __shared__ float scale_s;
    if (tid == 0) {
        float a = 0.0f, c2 = 0.0f;
        #pragma unroll
        for (int w = 0; w < 8; ++w) { a += sm[0][w]; c2 += sm[1][w]; }
        scale_s = sqrtf(a) / sqrtf(c2) + 1e-6f;
    }
    __syncthreads();
    float scale = scale_s;
    for (int i = tid; i < EMBD; i += 256)
        qout[(size_t)b * EMBD + i] = (__bf16)(x[i] + scale * rv[i]);
}

// ---------------------------------------------------------------------------
// GEMM3: out[4096][12288] = quantized_bf16 @ W_out + b_out   (f32 accumulate)
// Grid (256, 24): 16 rows x 512 cols per block; K = 512.
// ---------------------------------------------------------------------------
__global__ __launch_bounds__(256) void gemm3_out(const __bf16* __restrict__ qA,  // [4096][512]
                                                 const __bf16* __restrict__ WoT, // [12288][512]
                                                 const float* __restrict__ bout, // [12288]
                                                 float* __restrict__ out) {      // [4096][12288]
    const int tid  = threadIdx.x;
    const int wave = tid >> 5;
    const int lane = tid & 31;
    const int half = lane >> 4;
    const int r    = lane & 15;
    const int r0    = blockIdx.x * 16;
    const int nbase = blockIdx.y * 512 + wave * 64;

    v8f c[4] = {};
    const __bf16* ap = qA + (size_t)(r0 + r) * EMBD + half * 8;

    #pragma unroll 4
    for (int k0 = 0; k0 < EMBD; k0 += 32) {
        bf16x8 alo = *(const bf16x8*)(ap + k0);
        bf16x8 ahi = *(const bf16x8*)(ap + k0 + 16);
        bf16x16 a;
        #pragma unroll
        for (int i = 0; i < 8; ++i) { a[i] = alo[i]; a[8 + i] = ahi[i]; }
        #pragma unroll
        for (int t = 0; t < 4; ++t) {
            const __bf16* bp = WoT + (size_t)(nbase + t * 16 + r) * EMBD + k0 + half * 8;
            bf16x8 blo = *(const bf16x8*)(bp);
            bf16x8 bhi = *(const bf16x8*)(bp + 16);
            bf16x16 b;
            #pragma unroll
            for (int i = 0; i < 8; ++i) { b[i] = blo[i]; b[8 + i] = bhi[i]; }
            c[t] = __builtin_amdgcn_wmma_f32_16x16x32_bf16(false, a, false, b,
                                                           (short)0, c[t], false, false);
        }
    }
    #pragma unroll
    for (int t = 0; t < 4; ++t) {
        int col = nbase + t * 16 + r;
        float bb = bout[col];
        #pragma unroll
        for (int v = 0; v < 8; ++v)
            out[(size_t)(r0 + v + half * 8) * IN_DIM + col] = c[t][v] + bb;
    }
}

// ---------------------------------------------------------------------------
// Launch
// ---------------------------------------------------------------------------
extern "C" void kernel_launch(void* const* d_in, const int* in_sizes, int n_in,
                              void* d_out, int out_size, void* d_ws, size_t ws_size,
                              hipStream_t stream) {
    const float* img1  = (const float*)d_in[0];
    const float* img2  = (const float*)d_in[1];
    const float* randv = (const float*)d_in[2];
    const float* Win   = (const float*)d_in[3];
    // d_in[4] = b_in : cancels in x1 - x2, unused
    const float* Wout  = (const float*)d_in[5];
    const float* bout  = (const float*)d_in[6];
    const float* book  = (const float*)d_in[7];
    float* out = (float*)d_out;

    char* ws = (char*)d_ws;
    float*  inputf = (float*)(ws + OFF_INPUT);
    __bf16* WinT   = (__bf16*)(ws + OFF_WINT);
    __bf16* bookb  = (__bf16*)(ws + OFF_BOOK);
    float*  bnorm  = (float*)(ws + OFF_BNORM);
    int*    indices= (int*)(ws + OFF_IDX);
    __bf16* quantb = (__bf16*)(ws + OFF_QNT);
    __bf16* WoT    = (__bf16*)(ws + OFF_WOT);

    // Prep (independent, serialized on stream)
    prep_WinT <<<dim3(IN_DIM / 256, EMBD),   256, 0, stream>>>(Win, WinT);
    prep_WoutT<<<dim3(EMBD / 256, IN_DIM),   256, 0, stream>>>(Wout, WoT);
    prep_book <<<dim3(KBOOK),                256, 0, stream>>>(book, bookb, bnorm);

    // GEMM1: input = (img1 - img2) @ W_in   (b_in cancels)
    gemm1_in<<<dim3(BATCH / 16), 256, 0, stream>>>(img1, img2, WinT, inputf);

    // Fused distance + argmin over the codebook
    score_argmin<<<dim3(BATCH / 16), 256, 0, stream>>>(inputf, bookb, bnorm, indices);

    // Quantize epilogue -> bf16 activations
    quantize_rows<<<dim3(BATCH), 256, 0, stream>>>(inputf, book, randv, indices, quantb);

    // GEMM3: out = quantized @ W_out + b_out
    gemm3_out<<<dim3(BATCH / 16, IN_DIM / 512), 256, 0, stream>>>(quantb, WoT, bout, out);
}
